// Block_with_lora_27384711480006
// MI455X (gfx1250) — matrix operand order repaired and checked by reading, not compiled, not run on hardware
//
#include <hip/hip_runtime.h>
#include <hip/hip_bf16.h>
#include <math.h>

// ---------------------------------------------------------------------------
// MI455X (gfx1250) implementation of a GPT block with LoRA.
//   * All projections run through v_wmma_f32_16x16x32_bf16 (fp32->bf16 data,
//     fp32 accumulate). LoRA (rank 16) is folded into effective weights:
//     W_eff = W + (1/16) * LB @ A  -- exact, removes the skinny GEMMs.
//   * Attention is flash-style with WMMA for QK^T and PV, fp32 online softmax.
//   * Residual stream / LN / softmax stay fp32.
//   * Global->LDS tile staging uses CDNA5 async loads (ASYNCcnt) when the
//     toolchain exposes the builtin; falls back to reg-staged copies.
// Workspace layout (~128 MB): bf16 weights (32MB), fp32 residual (16MB),
// bf16 activations (80MB).
// ---------------------------------------------------------------------------

typedef __bf16 bf16;
typedef bf16  v16bf __attribute__((ext_vector_type(16)));
typedef float v8f   __attribute__((ext_vector_type(8)));
typedef unsigned int u32x4 __attribute__((ext_vector_type(4)));

#if defined(__has_builtin)
#if __has_builtin(__builtin_amdgcn_global_load_async_to_lds_b128)
#define USE_ASYNC_LDS 1
#endif
#endif
#ifndef USE_ASYNC_LDS
#define USE_ASYNC_LDS 0
#endif

// 16-byte global -> LDS copy.  On CDNA5 this lowers to
// global_load_async_to_lds_b128 (data bypasses VGPRs, tracked by ASYNCcnt).
__device__ __forceinline__ void copy16_to_lds(bf16* lds_dst, const bf16* gsrc) {
#if USE_ASYNC_LDS
  // Builtin signature (from toolchain): (v4i addrspace(1)*, v4i addrspace(3)*,
  // imm offset, imm cpol) with v4i = int __attribute__((vector_size(16))).
  typedef int v4i_gld __attribute__((vector_size(16)));
  typedef __attribute__((address_space(1))) v4i_gld* gp_t;
  typedef __attribute__((address_space(3))) v4i_gld* lp_t;
  gp_t g = (gp_t)(unsigned long long)(const void*)gsrc;
  lp_t l = (lp_t)(unsigned int)(unsigned long long)(void*)lds_dst;
  __builtin_amdgcn_global_load_async_to_lds_b128(g, l, 0, 0);
#else
  *(u32x4*)lds_dst = *(const u32x4*)gsrc;
#endif
}

// Wait for outstanding async global->LDS transfers (ASYNCcnt), which the
// workgroup barrier does NOT cover.
__device__ __forceinline__ void async_wait() {
#if USE_ASYNC_LDS
#if defined(__has_builtin) && __has_builtin(__builtin_amdgcn_s_wait_asynccnt)
  __builtin_amdgcn_s_wait_asynccnt(0);
#else
  asm volatile("s_wait_asynccnt 0x0" ::: "memory");
#endif
#endif
}

__device__ __forceinline__ v8f wmma_bf16(v16bf a, v16bf b, v8f c) {
  // (neg_a, A, neg_b, B, c_mod, C, reuse_a, reuse_b)
  return __builtin_amdgcn_wmma_f32_16x16x32_bf16(false, a, false, b, (short)0, c,
                                                 false, false);
}

// A-operand fragment (16x32 bf16). Layout: lanes 0-15 hold row M=lane,
// K=0..15; lanes 16-31 hold row M=lane-16, K=16..31.
// 'row' must point at (this lane's row, K-offset 0) with contiguous K.
__device__ __forceinline__ v16bf load_a32(const bf16* row) {
  const int hi = (threadIdx.x >> 4) & 1;
  union { v16bf v; u32x4 q[2]; } f;
  f.q[0] = *(const u32x4*)(row + hi * 16);
  f.q[1] = *(const u32x4*)(row + hi * 16 + 8);
  return f.v;
}

// B-operand fragment (32x16 bf16): lane holds column n = lane%16; lanes 0-15
// hold K=0..15, lanes 16-31 hold K=16..31.  'p' points at the 16 contiguous
// K values for this lane's (column, K-half).
__device__ __forceinline__ v16bf load_b32(const bf16* p) {
  union { v16bf v; u32x4 q[2]; } f;
  f.q[0] = *(const u32x4*)(p);
  f.q[1] = *(const u32x4*)(p + 8);
  return f.v;
}

// ---------------------------------------------------------------------------
// Effective weight: out[o,i] = bf16( W[o,i] + (1/16) * sum_r lb[o,r]*a[r,i] )
// ---------------------------------------------------------------------------
__global__ void fold_w_k(const float* __restrict__ W, const float* __restrict__ lb,
                         const float* __restrict__ a, bf16* __restrict__ out,
                         int OUTD, int IND) {
  int i = blockIdx.x * 256 + threadIdx.x;
  if (i >= OUTD * IND) return;
  int orow = i / IND;
  int col  = i - orow * IND;
  float val = W[i];
  if (lb != nullptr) {
    float s = 0.f;
#pragma unroll
    for (int r = 0; r < 16; ++r) s += lb[orow * 16 + r] * a[r * IND + col];
    val += s * (1.0f / 16.0f);
  }
  out[i] = (bf16)val;
}

__global__ void cvt_bf16_k(const float* __restrict__ in, bf16* __restrict__ out,
                           int n) {
  int i = blockIdx.x * 256 + threadIdx.x;
  if (i < n) out[i] = (bf16)in[i];
}

// ---------------------------------------------------------------------------
// Row LayerNorm (fp32 in) -> bf16 out.  One 256-thread block per row (C=1024).
// ---------------------------------------------------------------------------
__global__ __launch_bounds__(256)
void ln_k(const float* __restrict__ x, const float* __restrict__ g,
          const float* __restrict__ b, bf16* __restrict__ out, int C) {
  const int row = blockIdx.x;
  const float* xr = x + (size_t)row * C;
  const int tid = threadIdx.x;
  float s = 0.f, ss = 0.f;
  for (int i = tid; i < C; i += 256) { float v = xr[i]; s += v; ss += v * v; }
#pragma unroll
  for (int d = 16; d >= 1; d >>= 1) { s += __shfl_xor(s, d); ss += __shfl_xor(ss, d); }
  __shared__ float rs[8], rss[8];
  if ((tid & 31) == 0) { rs[tid >> 5] = s; rss[tid >> 5] = ss; }
  __syncthreads();
  s = 0.f; ss = 0.f;
  for (int w = 0; w < 8; ++w) { s += rs[w]; ss += rss[w]; }
  const float mean = s / C;
  const float var  = ss / C - mean * mean;
  const float rstd = rsqrtf(var + 1e-5f);
  bf16* orow = out + (size_t)row * C;
  for (int i = tid; i < C; i += 256)
    orow[i] = (bf16)((xr[i] - mean) * rstd * g[i] + b[i]);
}

// ---------------------------------------------------------------------------
// Tiled bf16 GEMM: out[M,N] = act( A[M,K] @ W[N,K]^T + bias ) (+ residual)
// Block tile 128x128, BK=32, 256 threads = 8 waves (2 row-bands x 4 col-bands),
// each wave: 4x2 = 8 v_wmma_f32_16x16x32_bf16 per K step.
// OUTF32: 0 -> bf16 out; 1 -> fp32 out with fp32 residual add.
// ---------------------------------------------------------------------------
template <int OUTF32, int GELU>
__global__ __launch_bounds__(256)
void gemm_bf16_k(const bf16* __restrict__ A, const bf16* __restrict__ W,
                 const float* __restrict__ bias, const float* __restrict__ res,
                 void* __restrict__ outp, int M, int N, int K) {
  __shared__ __align__(16) bf16 As[128 * 32];
  __shared__ __align__(16) bf16 Ws[128 * 32];
  const int tid  = threadIdx.x;
  const int lane = tid & 31;
  const int r16  = lane & 15;
  const int hi   = lane >> 4;
  const int wid  = tid >> 5;   // 0..7
  const int wr   = wid >> 2;   // 0..1  (64-row band)
  const int wc   = wid & 3;    // 0..3  (32-col band)
  const int bm   = blockIdx.y * 128;
  const int bn   = blockIdx.x * 128;

  v8f acc[4][2];
#pragma unroll
  for (int i = 0; i < 4; ++i)
#pragma unroll
    for (int j = 0; j < 2; ++j)
#pragma unroll
      for (int e = 0; e < 8; ++e) acc[i][j][e] = 0.f;

  for (int kb = 0; kb < K; kb += 32) {
#pragma unroll
    for (int s = 0; s < 2; ++s) {
      int slot = tid + s * 256;        // 512 uint4 slots per 128x32 tile
      int row  = slot >> 2;
      int c8   = (slot & 3) * 8;
      copy16_to_lds(&As[row * 32 + c8], A + (size_t)(bm + row) * K + kb + c8);
      copy16_to_lds(&Ws[row * 32 + c8], W + (size_t)(bn + row) * K + kb + c8);
    }
    async_wait();
    __syncthreads();
    v16bf bfr[2];
    bfr[0] = load_b32(&Ws[(wc * 32 + r16) * 32 + hi * 16]);
    bfr[1] = load_b32(&Ws[(wc * 32 + 16 + r16) * 32 + hi * 16]);
#pragma unroll
    for (int mt = 0; mt < 4; ++mt) {
      v16bf af = load_a32(&As[(wr * 64 + mt * 16 + r16) * 32]);
      acc[mt][0] = wmma_bf16(af, bfr[0], acc[mt][0]);
      acc[mt][1] = wmma_bf16(af, bfr[1], acc[mt][1]);
    }
    __syncthreads();
  }

  // Epilogue. C/D layout: lane holds column n=lane%16; VGPR j holds row
  // j + 8*(lane/16).
#pragma unroll
  for (int mt = 0; mt < 4; ++mt) {
#pragma unroll
    for (int nt = 0; nt < 2; ++nt) {
      const int col  = bn + wc * 32 + nt * 16 + r16;
      const float bc = bias[col];
#pragma unroll
      for (int j = 0; j < 8; ++j) {
        const int row = bm + wr * 64 + mt * 16 + hi * 8 + j;
        float v = acc[mt][nt][j] + bc;
        if (GELU) {
          float u = v;
          v = 0.5f * u * (1.0f + tanhf(0.7978845608028654f * (u + 0.044715f * u * u * u)));
        }
        if (OUTF32) {
          v += res[(size_t)row * N + col];
          ((float*)outp)[(size_t)row * N + col] = v;
        } else {
          ((bf16*)outp)[(size_t)row * N + col] = (bf16)v;
        }
      }
    }
  }
}

// ---------------------------------------------------------------------------
// Flash attention (causal / lower-triangular mask, D=64).
// Block = 128 threads (4 waves), each wave owns 16 q rows; k/v tiles of 64.
// QK^T: 8 WMMA per tile; PV: 8 WMMA per tile. Online softmax in fp32 with
// half-wave shfl_xor row reductions; P restaged via LDS into A-frag layout.
// Strides let the same kernel serve self- and cross-attention.
// ---------------------------------------------------------------------------
__global__ __launch_bounds__(128)
void attn_k(const bf16* __restrict__ q, const bf16* __restrict__ k,
            const bf16* __restrict__ v, bf16* __restrict__ o,
            int qs, int ks, int vs, int os, int T, int S, int H) {
  __shared__ __align__(16) bf16 Ks[64 * 64];
  __shared__ __align__(16) bf16 Vt[64 * 64];          // transposed: [d][s]
  __shared__ __align__(16) bf16 Ps[4][16 * 64];       // per-wave P staging

  const int tid  = threadIdx.x;
  const int lane = tid & 31;
  const int wid  = tid >> 5;     // 0..3
  const int r16  = lane & 15;
  const int hi   = lane >> 4;
  const int bh   = blockIdx.y;
  const int b    = bh / H, h = bh % H;
  const bf16* qp = q + (size_t)b * T * qs + h * 64;
  const bf16* kp = k + (size_t)b * S * ks + h * 64;
  const bf16* vp = v + (size_t)b * S * vs + h * 64;
  bf16* op       = o + (size_t)b * T * os + h * 64;
  const int q0   = blockIdx.x * 64 + wid * 16;

  // Q fragments (16 rows x 64) -> two 16x32 A-frags, loaded once from global.
  v16bf qa[2];
  {
    const bf16* row = qp + (size_t)(q0 + r16) * qs;
    qa[0] = load_a32(row);
    qa[1] = load_a32(row + 32);
  }

  float m_i[8], l_i[8];
  v8f oacc[4];
#pragma unroll
  for (int j = 0; j < 8; ++j) { m_i[j] = -1e30f; l_i[j] = 0.f; }
#pragma unroll
  for (int dt = 0; dt < 4; ++dt)
#pragma unroll
    for (int j = 0; j < 8; ++j) oacc[dt][j] = 0.f;

  const int kend = blockIdx.x * 64 + 64;   // causal: only tiles with keys <= row max
  for (int s0 = 0; s0 < kend; s0 += 64) {
    // K tile: rows contiguous (for B-frags of QK^T, contraction = d);
    // staged with async global->LDS loads.
    for (int it = tid; it < 512; it += 128) {
      int r = it >> 3, c8 = (it & 7) * 8;
      copy16_to_lds(&Ks[r * 64 + c8], kp + (size_t)(s0 + r) * ks + c8);
    }
    // V tile transposed (for B-frags of PV, contraction = s)
    for (int it = tid; it < 1024; it += 128) {
      int r = it >> 4, c4 = (it & 15) * 4;
      const bf16* src = vp + (size_t)(s0 + r) * vs + c4;
      Vt[(c4 + 0) * 64 + r] = src[0];
      Vt[(c4 + 1) * 64 + r] = src[1];
      Vt[(c4 + 2) * 64 + r] = src[2];
      Vt[(c4 + 3) * 64 + r] = src[3];
    }
    async_wait();
    __syncthreads();

    // S = Q @ K^T  (16 x 64 per wave, 4 column sub-tiles)
    v8f sacc[4];
#pragma unroll
    for (int st = 0; st < 4; ++st) {
#pragma unroll
      for (int j = 0; j < 8; ++j) sacc[st][j] = 0.f;
      const bf16* krow = &Ks[(st * 16 + r16) * 64 + hi * 16];
      sacc[st] = wmma_bf16(qa[0], load_b32(krow), sacc[st]);
      sacc[st] = wmma_bf16(qa[1], load_b32(krow + 32), sacc[st]);
    }

    // online softmax; row owned by (hi, VGPR j): trow = q0 + hi*8 + j
#pragma unroll
    for (int j = 0; j < 8; ++j) {
      const int trow = q0 + hi * 8 + j;
      float mx = m_i[j];
#pragma unroll
      for (int st = 0; st < 4; ++st) {
        int sg = s0 + st * 16 + r16;
        float val = sacc[st][j] * 0.125f;          // 1/sqrt(64)
        if (sg > trow) val = -1e30f;               // causal / tri(T,S) mask
        sacc[st][j] = val;
        mx = fmaxf(mx, val);
      }
#pragma unroll
      for (int d2 = 1; d2 < 16; d2 <<= 1) mx = fmaxf(mx, __shfl_xor(mx, d2));
      const float alpha = __expf(m_i[j] - mx);
      float rsum = 0.f;
#pragma unroll
      for (int st = 0; st < 4; ++st) {
        float p = __expf(sacc[st][j] - mx);
        sacc[st][j] = p;
        rsum += p;
      }
#pragma unroll
      for (int d2 = 1; d2 < 16; d2 <<= 1) rsum += __shfl_xor(rsum, d2);
      m_i[j] = mx;
      l_i[j] = l_i[j] * alpha + rsum;
#pragma unroll
      for (int dt = 0; dt < 4; ++dt) oacc[dt][j] *= alpha;
    }

    // P (C-layout) -> LDS row-major 16x64 so it can be reloaded as A-frags
    bf16* pp = &Ps[wid][0];
#pragma unroll
    for (int st = 0; st < 4; ++st)
#pragma unroll
      for (int j = 0; j < 8; ++j)
        pp[(hi * 8 + j) * 64 + st * 16 + r16] = (bf16)sacc[st][j];
    __syncthreads();

    v16bf pa0 = load_a32(pp + r16 * 64);
    v16bf pa1 = load_a32(pp + r16 * 64 + 32);
#pragma unroll
    for (int dt = 0; dt < 4; ++dt) {
      const bf16* vrow = &Vt[(dt * 16 + r16) * 64 + hi * 16];
      oacc[dt] = wmma_bf16(pa0, load_b32(vrow), oacc[dt]);
      oacc[dt] = wmma_bf16(pa1, load_b32(vrow + 32), oacc[dt]);
    }
    __syncthreads();   // protect Ks/Vt/Ps before next tile
  }

#pragma unroll
  for (int dt = 0; dt < 4; ++dt) {
    const int col = dt * 16 + r16;
#pragma unroll
    for (int j = 0; j < 8; ++j) {
      const int row = q0 + hi * 8 + j;
      op[(size_t)row * os + col] = (bf16)(oacc[dt][j] / l_i[j]);
    }
  }
}

// ---------------------------------------------------------------------------
// Launch
// ---------------------------------------------------------------------------
extern "C" void kernel_launch(void* const* d_in, const int* in_sizes, int n_in,
                              void* d_out, int out_size, void* d_ws, size_t ws_size,
                              hipStream_t stream) {
  (void)in_sizes; (void)n_in; (void)out_size; (void)ws_size;
  constexpr int B = 4, T = 1024, S = 1024, C = 1024, H = 16;
  constexpr int M = B * T;  // 4096 tokens

  const float* x       = (const float*)d_in[0];
  const float* feat    = (const float*)d_in[1];
  const float* ln1_g   = (const float*)d_in[2];
  const float* ln1_b   = (const float*)d_in[3];
  const float* ln2_g   = (const float*)d_in[4];
  const float* ln2_b   = (const float*)d_in[5];
  const float* qkv_w   = (const float*)d_in[6];
  const float* qkv_b   = (const float*)d_in[7];
  const float* qkv_a   = (const float*)d_in[8];
  const float* qkv_lb  = (const float*)d_in[9];
  const float* sap_w   = (const float*)d_in[10];
  const float* sap_b   = (const float*)d_in[11];
  const float* sap_a   = (const float*)d_in[12];
  const float* sap_lb  = (const float*)d_in[13];
  const float* caq_w   = (const float*)d_in[14];
  const float* caq_b   = (const float*)d_in[15];
  const float* caq_a   = (const float*)d_in[16];
  const float* caq_lb  = (const float*)d_in[17];
  const float* cakv_w  = (const float*)d_in[18];
  const float* cakv_b  = (const float*)d_in[19];
  const float* cakv_a  = (const float*)d_in[20];
  const float* cakv_lb = (const float*)d_in[21];
  const float* cap_w   = (const float*)d_in[22];
  const float* cap_b   = (const float*)d_in[23];
  const float* cap_a   = (const float*)d_in[24];
  const float* cap_lb  = (const float*)d_in[25];
  const float* fc_w    = (const float*)d_in[26];
  const float* fc_b    = (const float*)d_in[27];
  const float* pr_w    = (const float*)d_in[28];
  const float* pr_b    = (const float*)d_in[29];

  char* ws = (char*)d_ws;
  size_t off = 0;
  auto take = [&](size_t bytes) {
    void* p = ws + off;
    off += (bytes + 255) & ~(size_t)255;
    return p;
  };
  bf16*  w_qkv  = (bf16*)take((size_t)3 * C * C * 2);
  bf16*  w_sap  = (bf16*)take((size_t)C * C * 2);
  bf16*  w_caq  = (bf16*)take((size_t)C * C * 2);
  bf16*  w_cakv = (bf16*)take((size_t)2 * C * C * 2);
  bf16*  w_cap  = (bf16*)take((size_t)C * C * 2);
  bf16*  w_fc   = (bf16*)take((size_t)4 * C * C * 2);
  bf16*  w_pr   = (bf16*)take((size_t)4 * C * C * 2);
  float* xres   = (float*)take((size_t)M * C * 4);
  bf16*  hbf    = (bf16*)take((size_t)M * C * 2);
  bf16*  big    = (bf16*)take((size_t)M * 4 * C * 2);  // qkv (3C) then MLP hidden (4C)
  bf16*  obuf   = (bf16*)take((size_t)M * C * 2);
  bf16*  featbf = (bf16*)take((size_t)M * C * 2);
  bf16*  kvbuf  = (bf16*)take((size_t)M * 2 * C * 2);
  bf16*  q2buf  = (bf16*)take((size_t)M * C * 2);

  auto fold = [&](const float* W, const float* lb, const float* a, bf16* out,
                  int OUTD, int IND) {
    int n = OUTD * IND;
    fold_w_k<<<(n + 255) / 256, 256, 0, stream>>>(W, lb, a, out, OUTD, IND);
  };
  fold(qkv_w,  qkv_lb,  qkv_a,  w_qkv,  3 * C, C);
  fold(sap_w,  sap_lb,  sap_a,  w_sap,  C, C);
  fold(caq_w,  caq_lb,  caq_a,  w_caq,  C, C);
  fold(cakv_w, cakv_lb, cakv_a, w_cakv, 2 * C, C);
  fold(cap_w,  cap_lb,  cap_a,  w_cap,  C, C);
  fold(fc_w, nullptr, nullptr, w_fc, 4 * C, C);
  fold(pr_w, nullptr, nullptr, w_pr, C, 4 * C);

  cvt_bf16_k<<<(M * C + 255) / 256, 256, 0, stream>>>(feat, featbf, M * C);

  const dim3 blk(256);
  const dim3 agrid(T / 64, B * H);

  // --- causal self-attention with LoRA ---
  ln_k<<<M, 256, 0, stream>>>(x, ln1_g, ln1_b, hbf, C);
  gemm_bf16_k<0, 0><<<dim3(3 * C / 128, M / 128), blk, 0, stream>>>(
      hbf, w_qkv, qkv_b, nullptr, big, M, 3 * C, C);
  attn_k<<<agrid, 128, 0, stream>>>(big, big + C, big + 2 * C, obuf,
                                    3 * C, 3 * C, 3 * C, C, T, S, H);
  gemm_bf16_k<1, 0><<<dim3(C / 128, M / 128), blk, 0, stream>>>(
      obuf, w_sap, sap_b, x, xres, M, C, C);

  // --- cross-attention with LoRA ---
  ln_k<<<M, 256, 0, stream>>>(xres, ln1_g, ln1_b, hbf, C);
  gemm_bf16_k<0, 0><<<dim3(C / 128, M / 128), blk, 0, stream>>>(
      hbf, w_caq, caq_b, nullptr, q2buf, M, C, C);
  gemm_bf16_k<0, 0><<<dim3(2 * C / 128, M / 128), blk, 0, stream>>>(
      featbf, w_cakv, cakv_b, nullptr, kvbuf, M, 2 * C, C);
  attn_k<<<agrid, 128, 0, stream>>>(q2buf, kvbuf, kvbuf + C, obuf,
                                    C, 2 * C, 2 * C, C, T, S, H);
  gemm_bf16_k<1, 0><<<dim3(C / 128, M / 128), blk, 0, stream>>>(
      obuf, w_cap, cap_b, xres, xres, M, C, C);

  // --- MLP with tanh-GELU ---
  ln_k<<<M, 256, 0, stream>>>(xres, ln2_g, ln2_b, hbf, C);
  gemm_bf16_k<0, 1><<<dim3(4 * C / 128, M / 128), blk, 0, stream>>>(
      hbf, w_fc, fc_b, nullptr, big, M, 4 * C, C);
  gemm_bf16_k<1, 0><<<dim3(C / 128, M / 128), blk, 0, stream>>>(
      big, w_pr, pr_b, xres, (float*)d_out, M, C, 4 * C);
}
